// MultiHeadAttentionBlock_9904194585000
// MI455X (gfx1250) — matrix operand order repaired and checked by reading, not compiled, not run on hardware
//
#include <hip/hip_runtime.h>
#include <hip/hip_bf16.h>

typedef __attribute__((ext_vector_type(16))) _Float16 v16h;
typedef __attribute__((ext_vector_type(8)))  float    v8f;
typedef __attribute__((ext_vector_type(4)))  _Float16 h4;
typedef __attribute__((ext_vector_type(4)))  unsigned int u32x4;
typedef __attribute__((ext_vector_type(8)))  int          i32x8;
typedef __attribute__((ext_vector_type(4)))  int          i32x4;
typedef _Float16 half_t;

// ---- staging helpers: load 4 elements from global, store 4 halfs to LDS ----
__device__ __forceinline__ void load4cvt(half_t* dst, const float* src) {
    const float4 f = *(const float4*)src;
    h4 h;
    h.x = (half_t)f.x; h.y = (half_t)f.y; h.z = (half_t)f.z; h.w = (half_t)f.w;
    *(h4*)dst = h;
}
__device__ __forceinline__ void load4cvt(half_t* dst, const half_t* src) {
    *(h4*)dst = *(const h4*)src;
}

// ---- Tensor Data Mover: 2D f16 tile (32 k-elems wide x `rows`) -> LDS ----
// D# built per CDNA5 ISA. 6-arg builtin form (clang-23).
__device__ __forceinline__ void tdm_load_2d_f16(unsigned lds_byte_off,
                                                const void* gptr,
                                                int rows, int row_stride_elems)
{
    unsigned long long ga = (unsigned long long)gptr;
    u32x4 g0;
    g0[0] = 1u;                                               // count=1, user mode
    g0[1] = lds_byte_off;                                     // lds_addr
    g0[2] = (unsigned)ga;                                     // global_addr[31:0]
    g0[3] = (unsigned)((ga >> 32) & 0x1FFFFFFull) | (2u << 30); // ga[56:32] | type=2
    i32x8 g1;
    g1[0] = (1 << 16);                    // data_size=1 (2 bytes/elem)
    g1[1] = (32 & 0xFFFF) << 16;          // tensor_dim0 = 32 (lo16)
    g1[2] = ((rows & 0xFFFF) << 16);      // tensor_dim0 hi | tensor_dim1 lo16
    g1[3] = (32 << 16);                   // tensor_dim1 hi | tile_dim0 = 32
    g1[4] = rows & 0xFFFF;                // tile_dim1 = rows, tile_dim2 = 0
    g1[5] = row_stride_elems;             // tensor_dim0_stride lo32
    g1[6] = 0;                            // stride0 hi16 | stride1 lo16
    g1[7] = 0;                            // stride1 hi32
    const i32x4 z4 = {0, 0, 0, 0};
    const i32x8 z8 = {0, 0, 0, 0, 0, 0, 0, 0};
    __builtin_amdgcn_tensor_load_to_lds(g0, g1, z4, z4, z8, 0);
}

// =====================================================================
// Generic WMMA GEMM:  C[m,n] = alpha * sum_k A[m,k] * B[n,k]
// Block tile: 256(M) x 64(N), K-step 32, 256 threads = 8 wave32.
// Wave tile: 64x32 = 4x2 tiles of v_wmma_f32_16x16x32_f16 (8 wmma/kstep).
// Dynamic LDS only: A tile [0,16KB), B tile [16KB,20KB)  (launch with 20480B).
// Batched over blockIdx.z with offsets off = (z>>4)*s_hi + (z&15)*s_lo.
//
// USE_TDM (requires AT==BT==half_t): tiles staged by TENSOR_LOAD_TO_LDS
// issued from wave 0, synced with s_wait_tensorcnt + barrier.
//
// CMODE: 0 = f32 store (ldC), 1 = f16 store (ldC),
//        2 = f16 head-transposed store: out[((b*16+h)*64+nh)*1024 + l]
// =====================================================================
template <typename AT, typename BT, int CMODE, bool USE_TDM>
__global__ void __launch_bounds__(256)
wmma_gemm(const AT* __restrict__ A, const BT* __restrict__ B, void* __restrict__ C,
          int ldA, int ldB, int ldC, int K,
          long long sAh, long long sAl, long long sBh, long long sBl,
          long long sCh, long long sCl, float alpha)
{
    extern __shared__ half_t dynsmem[];
    half_t* As = dynsmem;                 // 256 x 32 f16 = 16 KB, LDS offset 0
    half_t* Bs = dynsmem + 256 * 32;      //  64 x 32 f16 =  4 KB, LDS offset 16384

    const int tid  = threadIdx.x;
    const int lane = tid & 31;
    const int wave = tid >> 5;           // 0..7
    const int mw   = (wave >> 1) * 64;   // 0,64,128,192
    const int nw   = (wave & 1) * 32;    // 0,32

    const int m0 = blockIdx.y * 256;
    const int n0 = blockIdx.x * 64;
    const int z  = blockIdx.z;
    const long long offA = (long long)(z >> 4) * sAh + (long long)(z & 15) * sAl;
    const long long offB = (long long)(z >> 4) * sBh + (long long)(z & 15) * sBl;
    const long long offC = (long long)(z >> 4) * sCh + (long long)(z & 15) * sCl;

    const v8f zero8 = {0.f, 0.f, 0.f, 0.f, 0.f, 0.f, 0.f, 0.f};
    v8f acc[4][2];
    #pragma unroll
    for (int mt = 0; mt < 4; ++mt) { acc[mt][0] = zero8; acc[mt][1] = zero8; }

    const int kb_a = (lane >> 4) << 3;   // 0 or 8
    const int kb_b = (lane >> 4) << 3;   // 0 or 8
    const int rsel = lane & 15;

    for (int k0 = 0; k0 < K; k0 += 32) {
        if constexpr (USE_TDM) {
            // ---- async tile staging via Tensor Data Mover (wave 0 only) ----
            if (wave == 0) {
                tdm_load_2d_f16(0u,     (const void*)(A + offA + (long long)m0 * ldA + k0),
                                256, ldA);
                tdm_load_2d_f16(16384u, (const void*)(B + offB + (long long)n0 * ldB + k0),
                                64, ldB);
                __builtin_amdgcn_s_wait_tensorcnt(0);
            }
            __syncthreads();
        } else {
            // ---- manual stage A tile: 256 rows x 32 k (cvt f32->f16) ----
            #pragma unroll
            for (int i = 0; i < 8; ++i) {
                int v   = tid + i * 256;        // 0..2047 vec4 slots
                int row = v >> 3;               // 0..255
                int col = (v & 7) << 2;         // 0..28
                load4cvt(&As[row * 32 + col],
                         A + offA + (long long)(m0 + row) * ldA + k0 + col);
            }
            // ---- stage B tile: 64 rows (n) x 32 k ----
            #pragma unroll
            for (int i = 0; i < 2; ++i) {
                int v   = tid + i * 256;        // 0..511
                int row = v >> 3;               // 0..63
                int col = (v & 7) << 2;
                load4cvt(&Bs[row * 32 + col],
                         B + offB + (long long)(n0 + row) * ldB + k0 + col);
            }
            __syncthreads();
        }

        // ---- fragment loads per documented CDNA5 WMMA VGPR layouts ----
        union F16x16 { v16h v; float4 f[2]; };
        F16x16 a[4], b[2];
        #pragma unroll
        for (int mt = 0; mt < 4; ++mt) {
            const half_t* p = As + (mw + mt * 16 + rsel) * 32;
            a[mt].f[0] = *(const float4*)(p + kb_a);        // K = kb..kb+7
            a[mt].f[1] = *(const float4*)(p + kb_a + 16);   // K = kb+16..kb+23
        }
        #pragma unroll
        for (int nt = 0; nt < 2; ++nt) {
            const half_t* p = Bs + (nw + nt * 16 + rsel) * 32 + kb_b;
            b[nt].f[0] = *(const float4*)(p);               // K = kb..kb+7
            b[nt].f[1] = *(const float4*)(p + 16);          // K = kb+16..kb+23
        }
        #pragma unroll
        for (int mt = 0; mt < 4; ++mt)
            #pragma unroll
            for (int nt = 0; nt < 2; ++nt)
                acc[mt][nt] = __builtin_amdgcn_wmma_f32_16x16x32_f16(
                    false, a[mt].v, false, b[nt].v,
                    (short)0, acc[mt][nt], false, false);
        __syncthreads();
    }

    // ---- epilogue: C/D layout is (m = v + 8*(lane>>4), n = lane&15) ----
    #pragma unroll
    for (int mt = 0; mt < 4; ++mt)
        #pragma unroll
        for (int nt = 0; nt < 2; ++nt)
            #pragma unroll
            for (int v = 0; v < 8; ++v) {
                float val = acc[mt][nt][v] * alpha;
                int mg = m0 + mw + mt * 16 + v + ((lane >> 4) << 3);
                int ng = n0 + nw + nt * 16 + (lane & 15);
                if (CMODE == 0) {
                    ((float*)C)[offC + (long long)mg * ldC + ng] = val;
                } else if (CMODE == 1) {
                    ((half_t*)C)[offC + (long long)mg * ldC + ng] = (half_t)val;
                } else {
                    // head-transposed V store: Vt[b,h,nh,l]
                    int bb = mg >> 10, ll = mg & 1023;
                    int hh = ng >> 6,  nh = ng & 63;
                    ((half_t*)C)[((long long)((bb << 4) + hh) * 64 + nh) * 1024 + ll] =
                        (half_t)val;
                }
            }
}

// =====================================================================
// Row softmax over 1024-wide rows, in place. 256 threads, float4/thread.
// =====================================================================
__global__ void __launch_bounds__(256)
softmax_rows(float* __restrict__ attn)
{
    const long long row = blockIdx.x;
    float* p = attn + row * 1024;
    const int tid = threadIdx.x;

    float4 v = ((const float4*)p)[tid];
    float m = fmaxf(fmaxf(v.x, v.y), fmaxf(v.z, v.w));
    #pragma unroll
    for (int off = 16; off > 0; off >>= 1) m = fmaxf(m, __shfl_xor(m, off, 32));

    __shared__ float redm[8];
    __shared__ float reds[8];
    if ((tid & 31) == 0) redm[tid >> 5] = m;
    __syncthreads();
    float bm = redm[0];
    #pragma unroll
    for (int i = 1; i < 8; ++i) bm = fmaxf(bm, redm[i]);

    v.x = __expf(v.x - bm); v.y = __expf(v.y - bm);
    v.z = __expf(v.z - bm); v.w = __expf(v.w - bm);
    float s = v.x + v.y + v.z + v.w;
    #pragma unroll
    for (int off = 16; off > 0; off >>= 1) s += __shfl_xor(s, off, 32);
    if ((tid & 31) == 0) reds[tid >> 5] = s;
    __syncthreads();
    float ts = 0.f;
    #pragma unroll
    for (int i = 0; i < 8; ++i) ts += reds[i];

    float inv = 1.0f / ts;
    v.x *= inv; v.y *= inv; v.z *= inv; v.w *= inv;
    ((float4*)p)[tid] = v;
}

// =====================================================================
// out = LayerNorm(y + bo + x) * gamma + beta, per 1024-wide row
// =====================================================================
__global__ void __launch_bounds__(256)
bias_residual_ln(const float* __restrict__ y, const float* __restrict__ x,
                 const float* __restrict__ bo, const float* __restrict__ gamma,
                 const float* __restrict__ beta, float* __restrict__ out)
{
    const long long row = blockIdx.x;
    const int tid = threadIdx.x;

    float4 yv = ((const float4*)(y + row * 1024))[tid];
    float4 xv = ((const float4*)(x + row * 1024))[tid];
    float4 bv = ((const float4*)bo)[tid];
    float4 t;
    t.x = yv.x + xv.x + bv.x;
    t.y = yv.y + xv.y + bv.y;
    t.z = yv.z + xv.z + bv.z;
    t.w = yv.w + xv.w + bv.w;

    float s  = t.x + t.y + t.z + t.w;
    float sq = t.x * t.x + t.y * t.y + t.z * t.z + t.w * t.w;
    #pragma unroll
    for (int off = 16; off > 0; off >>= 1) {
        s  += __shfl_xor(s,  off, 32);
        sq += __shfl_xor(sq, off, 32);
    }
    __shared__ float rs[8];
    __shared__ float rq[8];
    if ((tid & 31) == 0) { rs[tid >> 5] = s; rq[tid >> 5] = sq; }
    __syncthreads();
    float S = 0.f, Q = 0.f;
    #pragma unroll
    for (int i = 0; i < 8; ++i) { S += rs[i]; Q += rq[i]; }

    const float mean = S * (1.0f / 1024.0f);
    const float var  = Q * (1.0f / 1024.0f) - mean * mean;
    const float rstd = rsqrtf(var + 1e-5f);

    float4 g  = ((const float4*)gamma)[tid];
    float4 bb = ((const float4*)beta)[tid];
    float4 o;
    o.x = (t.x - mean) * rstd * g.x + bb.x;
    o.y = (t.y - mean) * rstd * g.y + bb.y;
    o.z = (t.z - mean) * rstd * g.z + bb.z;
    o.w = (t.w - mean) * rstd * g.w + bb.w;
    ((float4*)(out + row * 1024))[tid] = o;
}

// =====================================================================
extern "C" void kernel_launch(void* const* d_in, const int* in_sizes, int n_in,
                              void* d_out, int out_size, void* d_ws, size_t ws_size,
                              hipStream_t stream)
{
    (void)in_sizes; (void)n_in; (void)out_size; (void)ws_size;

    const float* x     = (const float*)d_in[0];
    const float* Wq    = (const float*)d_in[1];
    const float* Wk    = (const float*)d_in[2];
    const float* Wv    = (const float*)d_in[3];
    const float* Wo    = (const float*)d_in[4];
    const float* bo    = (const float*)d_in[5];
    const float* gamma = (const float*)d_in[6];
    const float* beta  = (const float*)d_in[7];

    const int B = 8, L = 1024, D = 1024, H = 16, Hd = 64;
    const long long M = (long long)B * L;          // 8192

    // workspace layout (48 MB total, with aliasing of dead buffers):
    //   [0,16M)  Q f16            -> later reused as context f16
    //   [16,32M) K f16            \__ later reused as y (f32, 32MB)
    //   [32,48M) Vt f16 (b,h,d,l) /
    char* ws = (char*)d_ws;
    half_t* Qw  = (half_t*)ws;
    half_t* Kw  = (half_t*)(ws + ((size_t)16 << 20));
    half_t* Vt  = (half_t*)(ws + ((size_t)32 << 20));
    half_t* Ctx = Qw;                                  // alias: Q dead by then
    float*  Yw  = (float*)(ws + ((size_t)16 << 20));   // alias: K,Vt dead by then

    float* outLN = (float*)d_out;
    float* attn  = (float*)d_out + (long long)B * L * D;   // second output

    dim3 blk(256);
    const size_t smem = (256 * 32 + 64 * 32) * sizeof(half_t);  // 20 KB

    // 1) Q = (x @ Wq^T) * (1/sqrt(Hd))  (scale folded into Q)
    wmma_gemm<float, float, 1, false><<<dim3(16, 32, 1), blk, smem, stream>>>(
        x, Wq, Qw, D, D, D, D, 0, 0, 0, 0, 0, 0, 0.125f);
    // 2) K = x @ Wk^T
    wmma_gemm<float, float, 1, false><<<dim3(16, 32, 1), blk, smem, stream>>>(
        x, Wk, Kw, D, D, D, D, 0, 0, 0, 0, 0, 0, 1.0f);
    // 3) Vt[b,h,d,l] = (x @ Wv^T) head-transposed
    wmma_gemm<float, float, 2, false><<<dim3(16, 32, 1), blk, smem, stream>>>(
        x, Wv, Vt, D, D, D, D, 0, 0, 0, 0, 0, 0, 1.0f);

    // 4) scores[b,h] = Q[b,:,h*64:] @ K[b,:,h*64:]^T  -> attn region (f32)
    //    pure f16 operands: staged by the Tensor Data Mover
    wmma_gemm<half_t, half_t, 0, true><<<dim3(16, 4, 128), blk, smem, stream>>>(
        Qw, Kw, attn, D, D, L, Hd,
        (long long)L * D, 64,
        (long long)L * D, 64,
        (long long)16 * L * L, (long long)L * L, 1.0f);

    // 5) softmax rows, in place -> final attn output
    softmax_rows<<<dim3((unsigned)(B * H * L)), blk, 0, stream>>>(attn);

    // 6) context[b,:,h*64+d] = attn[b,h] @ Vt[b,h]^T   (f16)
    wmma_gemm<float, half_t, 1, false><<<dim3(1, 4, 128), blk, smem, stream>>>(
        attn, Vt, Ctx, L, L, D, L,
        (long long)16 * L * L, (long long)L * L,
        (long long)H * Hd * L, (long long)Hd * L,
        (long long)L * D, 64, 1.0f);

    // 7) y = context @ Wo^T   (f32)
    wmma_gemm<half_t, float, 0, false><<<dim3(16, 32, 1), blk, smem, stream>>>(
        Ctx, Wo, Yw, D, D, D, D, 0, 0, 0, 0, 0, 0, 1.0f);

    // 8) out = LN(y + bo + x) * gamma + beta
    bias_residual_ln<<<dim3((unsigned)M), blk, 0, stream>>>(
        Yw, x, bo, gamma, beta, outLN);
}
